// MultiheadAttention_13340168421492
// MI455X (gfx1250) — compile-verified, matrix operand order checked by reading, not run on hardware
//
#include <hip/hip_runtime.h>

typedef __bf16 bf16;
typedef __attribute__((ext_vector_type(16))) __bf16 v16bf;
typedef __attribute__((ext_vector_type(8)))  float  v8f;

#define SEQ    1024
#define DMODEL 1024
#define HEADS  16
#define DK     64

// Per CDNA5 ISA 16-bit A/B fragment layout (16x32, MxK):
// lanes 0-15 hold K = {0..7, 16..23}; lanes 16-31 hold K = {8..15, 24..31}
__device__ __forceinline__ int kmap(int i, int laneHi) {
  return ((i < 8) ? i : (i + 8)) + (laneHi << 3);
}

// Async global->LDS copy (B128 chunks), GVS addressing. Per ISA
// CDNA5 ISA the 24-bit inst offset is added to BOTH the LDS dest and
// the global source, so one (ldsaddr, goff) pair serves all chunks. ASYNCcnt.
__device__ __forceinline__ void async_copy64(void* lds_dst, const void* gbase,
                                             unsigned goff) {
  unsigned ldsaddr = (unsigned)(size_t)lds_dst;
  asm volatile(
      "global_load_async_to_lds_b128 %0, %1, %2 offset:0\n\t"
      "global_load_async_to_lds_b128 %0, %1, %2 offset:16\n\t"
      "global_load_async_to_lds_b128 %0, %1, %2 offset:32\n\t"
      "global_load_async_to_lds_b128 %0, %1, %2 offset:48"
      :
      : "v"(ldsaddr), "v"(goff), "s"(gbase)
      : "memory");
}
__device__ __forceinline__ void async_copy32(void* lds_dst, const void* gbase,
                                             unsigned goff) {
  unsigned ldsaddr = (unsigned)(size_t)lds_dst;
  asm volatile(
      "global_load_async_to_lds_b128 %0, %1, %2 offset:0\n\t"
      "global_load_async_to_lds_b128 %0, %1, %2 offset:16"
      :
      : "v"(ldsaddr), "v"(goff), "s"(gbase)
      : "memory");
}
__device__ __forceinline__ void wait_async0() {
  asm volatile("s_wait_asynccnt 0x0" ::: "memory");
}

// ---------------------------------------------------------------------------
// 128x128-tile GEMM:  C[m][n] = sum_k X[m][k] * W[n][k] + bias[n]
// K staged in chunks of 64 (two WMMA K-steps per barrier pair).
// ---------------------------------------------------------------------------
template <bool IN_F32, bool PERMUTE_OUT, typename OutT>
__global__ __launch_bounds__(256) void gemm128(const void* __restrict__ Xv,
                                               const float* __restrict__ W,
                                               const float* __restrict__ bias,
                                               OutT* __restrict__ out) {
  constexpr int LDA = 72;  // padded bf16 row stride (144B) to spread LDS banks
  __shared__ bf16 As[128 * LDA];
  __shared__ bf16 Bs[128 * LDA];

  const int tid    = threadIdx.x;
  const int lane   = tid & 31;
  const int w      = tid >> 5;
  const int laneHi = lane >> 4;
  const int l16    = lane & 15;
  const int m0     = blockIdx.x * 128;
  const int n0     = blockIdx.y * 128;
  const int wave_m = w >> 2;  // 0..1 : 64 rows
  const int wave_n = w & 3;   // 0..3 : 32 cols

  v8f acc[4][2];
#pragma unroll
  for (int mt = 0; mt < 4; ++mt)
#pragma unroll
    for (int nt = 0; nt < 2; ++nt)
#pragma unroll
      for (int e = 0; e < 8; ++e) acc[mt][nt][e] = 0.0f;

  const int r  = tid >> 1;         // 0..127 staging row
  const int c0 = (tid & 1) * 32;   // 0 or 32

  for (int k0 = 0; k0 < DMODEL; k0 += 64) {
    __syncthreads();
    // ---- stage A tile (128 x 64) ----
    if (IN_F32) {
      const float* src = (const float*)Xv + (size_t)(m0 + r) * DMODEL + k0 + c0;
#pragma unroll
      for (int v = 0; v < 8; ++v) {
        float4 f = ((const float4*)src)[v];
        bf16* dst = &As[r * LDA + c0 + v * 4];
        dst[0] = (bf16)f.x; dst[1] = (bf16)f.y;
        dst[2] = (bf16)f.z; dst[3] = (bf16)f.w;
      }
    } else {
      const uint4* src =
          (const uint4*)((const bf16*)Xv + (size_t)(m0 + r) * DMODEL + k0 + c0);
      uint4* dst = (uint4*)&As[r * LDA + c0];
#pragma unroll
      for (int v = 0; v < 4; ++v) dst[v] = src[v];
    }
    // ---- stage B tile: rows n0..n0+127 of W (always fp32) ----
    {
      const float* src = W + (size_t)(n0 + r) * DMODEL + k0 + c0;
#pragma unroll
      for (int v = 0; v < 8; ++v) {
        float4 f = ((const float4*)src)[v];
        bf16* dst = &Bs[r * LDA + c0 + v * 4];
        dst[0] = (bf16)f.x; dst[1] = (bf16)f.y;
        dst[2] = (bf16)f.z; dst[3] = (bf16)f.w;
      }
    }
    __syncthreads();

#pragma unroll
    for (int kc = 0; kc < 2; ++kc) {
      v16bf a[4], bfr[2];
#pragma unroll
      for (int mt = 0; mt < 4; ++mt)
#pragma unroll
        for (int i = 0; i < 16; ++i)
          a[mt][i] = As[(wave_m * 64 + mt * 16 + l16) * LDA + kc * 32 +
                        kmap(i, laneHi)];
#pragma unroll
      for (int nt = 0; nt < 2; ++nt)
#pragma unroll
        for (int i = 0; i < 16; ++i)
          bfr[nt][i] = Bs[(wave_n * 32 + nt * 16 + l16) * LDA + kc * 32 +
                          kmap(i, laneHi)];

#pragma unroll
      for (int mt = 0; mt < 4; ++mt)
#pragma unroll
        for (int nt = 0; nt < 2; ++nt)
          acc[mt][nt] = __builtin_amdgcn_wmma_f32_16x16x32_bf16(
              false, a[mt], false, bfr[nt], (short)0, acc[mt][nt], false, false);
    }
  }

  // ---- epilogue ----
#pragma unroll
  for (int mt = 0; mt < 4; ++mt)
#pragma unroll
    for (int nt = 0; nt < 2; ++nt) {
      const int ng = n0 + wave_n * 32 + nt * 16 + l16;
      const float bv = bias[ng];
#pragma unroll
      for (int j = 0; j < 8; ++j) {
        const int mg = m0 + wave_m * 64 + mt * 16 + j + (laneHi << 3);
        const float val = acc[mt][nt][j] + bv;
        if (PERMUTE_OUT) {
          const int bi = mg >> 10, si = mg & 1023, h = ng >> 6, dd = ng & 63;
          out[(((size_t)(bi * HEADS + h)) * SEQ + si) * DK + dd] = (OutT)val;
        } else {
          // streamed final output: non-temporal, keep L2 for qkv/xw tiles
          __builtin_nontemporal_store((OutT)val,
                                      &out[(size_t)mg * DMODEL + ng]);
        }
      }
    }
}

// ---------------------------------------------------------------------------
// Attention: one block per (b*H, 64-row q tile). Scores stay in LDS (320KB WGP
// budget); Q/K/V staged with GLOBAL_LOAD_ASYNC_TO_LDS, double-buffered so the
// next block's DMA overlaps this block's WMMAs (and V block 0 overlaps the
// whole softmax).
// ---------------------------------------------------------------------------
#define SC_LD 1032  // padded fp32 score row stride
#define QT_LD 72    // padded bf16 strides
#define KV_LD 72
#define KVBUF_B (128 * KV_LD * 2)                // 18432 per staging buffer
#define SM_SC   (64 * SC_LD * 4)                 // 264192
#define SM_QT   (SM_SC + 64 * QT_LD * 2)         // +9216  = 273408 (kv buf 0)
#define SM_FM   (SM_QT + 2 * KVBUF_B)            // 310272
#define SM_TOT  (SM_FM + SEQ * 4 + 64 * 4 + 64 * 4 * 4)  // 315648 B < 320 KB

__global__ __launch_bounds__(256) void attn64(const bf16* __restrict__ qp,
                                              const bf16* __restrict__ kp,
                                              const bf16* __restrict__ vp,
                                              const int* __restrict__ mask,
                                              float* __restrict__ p_out,
                                              bf16* __restrict__ xw) {
  extern __shared__ char smem[];
  float* sc = (float*)smem;
  bf16*  qt = (bf16*)(smem + SM_SC);
  float* fm = (float*)(smem + SM_FM + 0);
  float* fq = (float*)(smem + SM_FM + SEQ * 4);
  float* red = (float*)(smem + SM_FM + SEQ * 4 + 64 * 4);

  const int tid = threadIdx.x;
  const int lane = tid & 31;
  const int w = tid >> 5;
  const int laneHi = lane >> 4;
  const int l16 = lane & 15;
  const int bh = blockIdx.x;       // b*16 + h
  const int b = bh >> 4, h = bh & 15;
  const int q0 = blockIdx.y * 64;
  const size_t bh_base = (size_t)bh * SEQ * DK;
  const int m_t = w & 3;           // wave's 16-row strip of the 64-row q tile

  const bf16* kbase = kp + bh_base;
  const bf16* vbase = vp + bh_base;
  const int st_row = tid >> 1, st_c = (tid & 1) * 32;  // 128x64 staging slot
  const unsigned st_lds = (unsigned)(st_row * KV_LD + st_c);

  // kv staging buffer i (i in {0,1}) -- computed, not a pointer array, so no
  // constant-initializer addrspacecast is emitted.
#define KVBUF(i) ((bf16*)(smem + SM_QT + (size_t)((i)&1) * KVBUF_B))

  // stage q tile (async) + prefetch K block 0 + masks
  {
    const int row = tid >> 2, c = (tid & 3) * 16;
    async_copy32(qt + row * QT_LD + c, qp + bh_base,
                 (unsigned)(((q0 + row) * DK + c) * 2));
  }
  async_copy64(KVBUF(0) + st_lds, kbase, (unsigned)((st_row * DK + st_c) * 2));
#pragma unroll
  for (int i = 0; i < 4; ++i)
    fm[tid + 256 * i] = (float)mask[b * SEQ + tid + 256 * i];
  if (tid < 64) fq[tid] = (float)mask[b * SEQ + q0 + tid];
  wait_async0();
  __syncthreads();

  // q A-fragments (reused across all 64 score column tiles)
  v16bf aq[2];
#pragma unroll
  for (int kc = 0; kc < 2; ++kc)
#pragma unroll
    for (int i = 0; i < 16; ++i)
      aq[kc][i] = qt[(m_t * 16 + l16) * QT_LD + kc * 32 + kmap(i, laneHi)];

  // ---- scores: S = Q K^T (double-buffered K staging) ----
  for (int ib = 0; ib < 8; ++ib) {
    const bf16* cur = KVBUF(ib);
    if (ib < 7)
      async_copy64(KVBUF(ib + 1) + st_lds, kbase,
                   (unsigned)((((ib + 1) * 128 + st_row) * DK + st_c) * 2));
    const int cb = ib * 128;
    const int ntg = (w >> 2) * 4;
#pragma unroll
    for (int t = 0; t < 4; ++t) {
      const int nt = ntg + t;
      v8f c8;
#pragma unroll
      for (int e = 0; e < 8; ++e) c8[e] = 0.0f;
#pragma unroll
      for (int kc = 0; kc < 2; ++kc) {
        v16bf bfr;
#pragma unroll
        for (int i = 0; i < 16; ++i)
          bfr[i] = cur[(nt * 16 + l16) * KV_LD + kc * 32 + kmap(i, laneHi)];
        c8 = __builtin_amdgcn_wmma_f32_16x16x32_bf16(false, aq[kc], false, bfr,
                                                     (short)0, c8, false, false);
      }
#pragma unroll
      for (int j = 0; j < 8; ++j)
        sc[(m_t * 16 + j + (laneHi << 3)) * SC_LD + cb + nt * 16 + l16] = c8[j];
    }
    wait_async0();   // next K block landed (and our score stores are DS-side)
    __syncthreads();
  }

  // prefetch V block 0: its DMA hides under the whole softmax
  async_copy64(KVBUF(0) + st_lds, vbase, (unsigned)((st_row * DK + st_c) * 2));

  // ---- masked softmax (scale=1/sqrt(64), pre-mask -1e9, post-mask multiply)
  {
    const float scale = 0.125f;
    const int row = tid >> 2, part = tid & 3, cbeg = part * 256;
    const float mq = fq[row];
    float lmax = -3.0e38f;
    for (int c = cbeg; c < cbeg + 256; ++c) {
      float s = sc[row * SC_LD + c] * scale;
      if (mq * fm[c] == 0.0f) s = -1.0e9f;
      lmax = fmaxf(lmax, s);
    }
    red[row * 4 + part] = lmax;
    __syncthreads();
    const float m = fmaxf(fmaxf(red[row * 4 + 0], red[row * 4 + 1]),
                          fmaxf(red[row * 4 + 2], red[row * 4 + 3]));
    __syncthreads();
    float lsum = 0.0f;
    for (int c = cbeg; c < cbeg + 256; ++c) {
      float s = sc[row * SC_LD + c] * scale;
      if (mq * fm[c] == 0.0f) s = -1.0e9f;
      lsum += __expf(s - m);
    }
    red[row * 4 + part] = lsum;
    __syncthreads();
    const float inv = 1.0f / (red[row * 4 + 0] + red[row * 4 + 1] +
                              red[row * 4 + 2] + red[row * 4 + 3]);
    float* prow = p_out + ((size_t)bh * SEQ + q0 + row) * SEQ;
    for (int c = cbeg; c < cbeg + 256; ++c) {
      const float mf = mq * fm[c];
      float s = sc[row * SC_LD + c] * scale;
      if (mf == 0.0f) s = -1.0e9f;
      const float p = __expf(s - m) * inv * mf;
      sc[row * SC_LD + c] = p;
      __builtin_nontemporal_store(p, &prow[c]);  // 512MB stream-once output
    }
  }
  wait_async0();   // V block 0 ready
  __syncthreads();

  // ---- X = P V (double-buffered V staging) ----
  const int nt0 = (w >> 2) * 2;
  v8f xc[2];
#pragma unroll
  for (int j = 0; j < 2; ++j)
#pragma unroll
    for (int e = 0; e < 8; ++e) xc[j][e] = 0.0f;

  for (int ib = 0; ib < 8; ++ib) {
    const bf16* cur = KVBUF(ib);
    if (ib < 7)
      async_copy64(KVBUF(ib + 1) + st_lds, vbase,
                   (unsigned)((((ib + 1) * 128 + st_row) * DK + st_c) * 2));
    const int kb = ib * 128;
#pragma unroll
    for (int kc4 = 0; kc4 < 4; ++kc4) {
      v16bf ap;
#pragma unroll
      for (int i = 0; i < 16; ++i)
        ap[i] = (bf16)sc[(m_t * 16 + l16) * SC_LD + kb + kc4 * 32 + kmap(i, laneHi)];
#pragma unroll
      for (int j = 0; j < 2; ++j) {
        v16bf bfr;
#pragma unroll
        for (int i = 0; i < 16; ++i)
          bfr[i] = cur[(kc4 * 32 + kmap(i, laneHi)) * KV_LD + (nt0 + j) * 16 + l16];
        xc[j] = __builtin_amdgcn_wmma_f32_16x16x32_bf16(false, ap, false, bfr,
                                                        (short)0, xc[j], false, false);
      }
    }
    wait_async0();
    __syncthreads();
  }
#undef KVBUF

  // store x in (b, s, H*dk) bf16 layout for the output projection
#pragma unroll
  for (int j = 0; j < 2; ++j) {
    const int nin = h * DK + (nt0 + j) * 16 + l16;
#pragma unroll
    for (int e = 0; e < 8; ++e) {
      const int srow = q0 + m_t * 16 + e + (laneHi << 3);
      xw[((size_t)b * SEQ + srow) * DMODEL + nin] = (bf16)xc[j][e];
    }
  }
}

// ---------------------------------------------------------------------------
extern "C" void kernel_launch(void* const* d_in, const int* in_sizes, int n_in,
                              void* d_out, int out_size, void* d_ws, size_t ws_size,
                              hipStream_t stream) {
  const float* q    = (const float*)d_in[0];
  const float* k    = (const float*)d_in[1];
  const float* v    = (const float*)d_in[2];
  const int*   mask = (const int*)d_in[3];
  const float* Wq   = (const float*)d_in[4];
  const float* bq   = (const float*)d_in[5];
  const float* Wo   = (const float*)d_in[6];
  const float* bo   = (const float*)d_in[7];

  float* out   = (float*)d_out;
  float* p_out = out + (size_t)8 * SEQ * DMODEL;  // p_attn region of d_out

  const size_t elems = (size_t)8 * SEQ * DMODEL;  // 8M elements per tensor
  bf16* qp = (bf16*)d_ws;
  bf16* kp = qp + elems;
  bf16* vp = kp + elems;
  bf16* xw = vp + elems;

  static_assert(SM_TOT <= 320 * 1024, "LDS budget");
  (void)hipFuncSetAttribute((const void*)attn64,
                            hipFuncAttributeMaxDynamicSharedMemorySize, SM_TOT);

  dim3 gg(8192 / 128, DMODEL / 128);  // (64, 8)
  // shared projection (reference applies Wq/bq to q, k AND v)
  gemm128<true, true, bf16><<<gg, 256, 0, stream>>>((const void*)q, Wq, bq, qp);
  gemm128<true, true, bf16><<<gg, 256, 0, stream>>>((const void*)k, Wq, bq, kp);
  gemm128<true, true, bf16><<<gg, 256, 0, stream>>>((const void*)v, Wq, bq, vp);

  attn64<<<dim3(8 * HEADS, SEQ / 64), 256, SM_TOT, stream>>>(qp, kp, vp, mask,
                                                             p_out, xw);

  gemm128<false, false, float><<<gg, 256, 0, stream>>>((const void*)xw, Wo, bo,
                                                       out);
}
